// UnidirectionalLM_35278861369607
// MI455X (gfx1250) — matrix operand order repaired and checked by reading, not compiled, not run on hardware
//
#include <hip/hip_runtime.h>
#include <hip/hip_bf16.h>

// Problem dims (compile-time constants from the reference)
#define B_  32
#define S_  2048
#define D_  512
#define L_  2
#define N3D (3 * D_)        // 1536
#define M_  (S_ * B_)       // 65536 rows of the GEMM

#define BM_ 128             // rows per workgroup (8 waves x 16)
#define BN_ 64              // cols per workgroup (4 WMMA tiles, shared W strip)
#define LDS_STRIDE 520      // 512 + 8 bf16 pad: lane stride = 260 dwords -> banks 4*lm, conflict-free b128

typedef __bf16 v16bf __attribute__((ext_vector_type(16)));
typedef float  v8f   __attribute__((ext_vector_type(8)));

// round-to-nearest-even fp32 -> bf16
static __device__ inline __bf16 f2bf(float f) {
    union { float f; unsigned u; } in; in.f = f;
    unsigned u = in.u;
    unsigned r = u + 0x7FFFu + ((u >> 16) & 1u);
    unsigned short hs = (unsigned short)(r >> 16);
    __bf16 o;
    __builtin_memcpy(&o, &hs, sizeof(o));
    return o;
}

// weights (L, D, 3D) fp32 -> Wt (L, 3D, D) bf16 (K-contiguous per output col)
__global__ void sru_wt_kernel(const float* __restrict__ W, __bf16* __restrict__ Wt) {
    long i = (long)blockIdx.x * blockDim.x + threadIdx.x;
    if (i >= (long)L_ * N3D * D_) return;
    int  k = (int)(i % D_);
    long r = i / D_;
    int  n = (int)(r % N3D);
    int  l = (int)(r / N3D);
    Wt[i] = f2bf(W[((long)l * D_ + k) * N3D + n]);
}

// x[s][b][d] = table[ids[b][s]][d]   (fp32 + bf16 copies)
__global__ void sru_embed_kernel(const int* __restrict__ ids,
                                 const float* __restrict__ table,
                                 float* __restrict__ x,
                                 __bf16* __restrict__ xb) {
    long i = (long)blockIdx.x * blockDim.x + threadIdx.x; // over S*B*D
    if (i >= (long)M_ * D_) return;
    int  d  = (int)(i % D_);
    long sb = i / D_;
    int  b  = (int)(sb % B_);
    int  s  = (int)(sb / B_);
    int  id = ids[(long)b * S_ + s];
    float v = table[(long)id * D_ + d];
    x[i]  = v;
    xb[i] = f2bf(v);
}

// U[m][n] = sum_k A[m][k] * Wt[n][k];  A: [M][512] bf16, Wt: [1536][512] bf16.
// Workgroup = 8 waves: BM=128 rows, BN=64 cols. The 64x512 W strip is staged in
// LDS once per workgroup (64 KB) and feeds all 8 waves' B fragments via
// ds_load_b128 -> 8x less L2 traffic on W. fp32 WMMA accumulation, bf16 U store.
__global__ __launch_bounds__(256)
void sru_gemm_kernel(const __bf16* __restrict__ A,
                     const __bf16* __restrict__ Wt,
                     __bf16* __restrict__ U) {
    __shared__ __bf16 Bs[BN_ * LDS_STRIDE];   // 66,560 B

    const int nblk = N3D / BN_;               // 24 N strips
    int mtblk = blockIdx.x / nblk;            // consecutive blocks share A rows (L2 reuse)
    int ntblk = blockIdx.x % nblk;
    int n0 = ntblk * BN_;

    // ---- cooperative stage of W strip [n0 .. n0+63][0..511] into padded LDS ----
    {
        const __bf16* wsrc = Wt + (long)n0 * D_;
        for (int idx = threadIdx.x; idx < BN_ * (D_ / 8); idx += blockDim.x) {
            int row  = idx / (D_ / 8);        // 0..63
            int col8 = idx % (D_ / 8);        // 16-byte chunk within the row
            *reinterpret_cast<uint4*>(&Bs[row * LDS_STRIDE + col8 * 8]) =
                *reinterpret_cast<const uint4*>(wsrc + (long)row * D_ + col8 * 8);
        }
    }
    __syncthreads();

    const int NT = BN_ / 16;                  // 4 WMMA tiles per wave
    int wave = threadIdx.x >> 5;
    int lane = threadIdx.x & 31;
    int m0 = mtblk * BM_ + wave * 16;
    int lm = lane & 15;   // M (A) / N (B,C,D) index within tile
    int lh = lane >> 4;   // half-wave select

    v8f acc[NT] = {};

    // A fragment: lane lm = row M; lh=0 -> K {0..7,16..23}, lh=1 -> K {8..15,24..31}
    const __bf16* arow = A + (long)(m0 + lm) * D_;
    for (int kk = 0; kk < D_; kk += 32) {
        union { v16bf v; uint4 q[2]; } af;
        af.q[0] = *reinterpret_cast<const uint4*>(arow + kk + lh * 8);
        af.q[1] = *reinterpret_cast<const uint4*>(arow + kk + 16 + lh * 8);
#pragma unroll
        for (int j = 0; j < NT; ++j) {
            // B fragment from LDS: lane lm = col N; lh=0 -> K {0..7,16..23}, lh=1 -> K {8..15,24..31}
            const __bf16* brow = &Bs[(j * 16 + lm) * LDS_STRIDE + kk + lh * 8];
            union { v16bf v; uint4 q[2]; } bfr;
            bfr.q[0] = reinterpret_cast<const uint4*>(brow)[0];
            bfr.q[1] = reinterpret_cast<const uint4*>(brow + 16)[0];
            acc[j] = __builtin_amdgcn_wmma_f32_16x16x32_bf16(
                false, af.v, false, bfr.v, (short)0, acc[j], false, false);
        }
    }

    // C/D layout: VGPR r, lanes 0-15 -> M=r, lanes 16-31 -> M=r+8; N = lane%16
#pragma unroll
    for (int j = 0; j < NT; ++j) {
#pragma unroll
        for (int r = 0; r < 8; ++r) {
            int mrow = m0 + r + lh * 8;
            U[(long)mrow * N3D + n0 + j * 16 + lm] = f2bf(acc[j][r]);
        }
    }
}

// Sequential SRU scan: one thread per (b, d); threads contiguous in d -> coalesced.
// Recurrence is nonlinear in c (c appears inside the sigmoids), so parallelism is
// fixed at B*D = 16384 lanes; hide HBM latency with software prefetch instead.
// Fused epilogue: optionally emits the bf16 copy of h (next layer's GEMM input),
// removing the standalone fp32->bf16 conversion pass (~200 MB of traffic).
__global__ __launch_bounds__(256)
void sru_scan_kernel(const __bf16* __restrict__ U,    // [S*B][3D] bf16
                     const float* __restrict__ x,     // [S][B][D]
                     const unsigned char* __restrict__ mask, // [B][S], nonzero = valid
                     const float* __restrict__ bias,  // [2D] this layer
                     const float* __restrict__ vsc,   // [2D] this layer
                     float* __restrict__ h,           // output (fp32)
                     __bf16* __restrict__ hb,         // optional bf16 copy (next GEMM A), may be null
                     int out_bsd)                     // 1 -> h[b][s][d], else h[s][b][d]
{
    const int PF = 8;   // prefetch distance (iterations)
    int t = blockIdx.x * blockDim.x + threadIdx.x;
    if (t >= B_ * D_) return;
    int d = t % D_;
    int b = t / D_;
    float bf = bias[d], br = bias[D_ + d];
    float vf = vsc[d],  vr = vsc[D_ + d];
    float c = 0.0f;
    const unsigned char* mrow = mask + (long)b * S_;
    for (int s = 0; s < S_; ++s) {
        long row = (long)s * B_ + b;
        const __bf16* u = U + row * N3D;
        if (s + PF < S_) {
            long prow = (long)(s + PF) * B_ + b;
            const __bf16* pu = U + prow * N3D;
            __builtin_prefetch(pu + d,          0, 1);   // global_prefetch_b8
            __builtin_prefetch(pu + D_ + d,     0, 1);
            __builtin_prefetch(pu + 2 * D_ + d, 0, 1);
            __builtin_prefetch(x + prow * D_ + d, 0, 1);
        }
        float u0 = (float)u[d];
        float u1 = (float)u[D_ + d];
        float u2 = (float)u[2 * D_ + d];
        float xt = x[row * D_ + d];
        float g1 = 1.0f / (1.0f + __expf(-(u1 + bf + c * vf)));
        float cn = (c - u0) * g1 + u0;
        float g2 = 1.0f / (1.0f + __expf(-(u2 + br + c * vr)));   // uses old c
        float hv = (tanhf(cn) - xt) * g2 + xt;
        bool pad = (mrow[s] == 0);
        c  = pad ? c : cn;
        hv = pad ? 0.0f : hv;
        long oidx = out_bsd ? ((long)b * S_ + s) * D_ + d : row * D_ + d;
        h[oidx] = hv;
        if (hb) hb[row * D_ + d] = f2bf(hv);   // fused convert for next layer's GEMM
    }
}

extern "C" void kernel_launch(void* const* d_in, const int* in_sizes, int n_in,
                              void* d_out, int out_size, void* d_ws, size_t ws_size,
                              hipStream_t stream) {
    const int*           ids   = (const int*)d_in[0];
    const unsigned char* mask  = (const unsigned char*)d_in[1];
    const float*         table = (const float*)d_in[2];
    const float*         W     = (const float*)d_in[3];
    const float*         bias  = (const float*)d_in[4];
    const float*         vs    = (const float*)d_in[5];
    float*               out   = (float*)d_out;

    char* ws = (char*)d_ws;
    size_t off = 0;
    float*  x0 = (float*)(ws + off);  off += (size_t)M_ * D_ * sizeof(float);
    float*  x1 = (float*)(ws + off);  off += (size_t)M_ * D_ * sizeof(float);
    __bf16* xb = (__bf16*)(ws + off); off += (size_t)M_ * D_ * sizeof(__bf16);
    __bf16* Wt = (__bf16*)(ws + off); off += (size_t)L_ * N3D * D_ * sizeof(__bf16);
    __bf16* U  = (__bf16*)(ws + off); off += (size_t)M_ * N3D * sizeof(__bf16);

    {   // weight transpose/convert (deterministic, re-run every call)
        long n = (long)L_ * N3D * D_;
        sru_wt_kernel<<<(unsigned)((n + 255) / 256), 256, 0, stream>>>(W, Wt);
    }
    {   // embedding gather (fp32 x + bf16 GEMM input in one pass)
        long n = (long)M_ * D_;
        sru_embed_kernel<<<(unsigned)((n + 255) / 256), 256, 0, stream>>>(ids, table, x0, xb);
    }

    float* xcur = x0;
    float* xnext = x1;
    for (int l = 0; l < L_; ++l) {
        int nblocks = (M_ / BM_) * (N3D / BN_);   // 512 * 24 = 12288 workgroups
        sru_gemm_kernel<<<nblocks, 256, 0, stream>>>(
            xb, Wt + (long)l * N3D * D_, U);

        int last = (l == L_ - 1);
        sru_scan_kernel<<<(B_ * D_ + 255) / 256, 256, 0, stream>>>(
            U, xcur, mask, bias + (long)l * 2 * D_, vs + (long)l * 2 * D_,
            last ? out : xnext,
            last ? (__bf16*)nullptr : xb,     // fused bf16 emit for next layer
            last);

        float* t = xcur; xcur = xnext; xnext = t;
    }
}
